// ProbSparseAttention_73040213835824
// MI455X (gfx1250) — matrix operand order repaired and checked by reading, not compiled, not run on hardware
//
#include <hip/hip_runtime.h>
#include <hip/hip_bf16.h>

typedef __attribute__((ext_vector_type(16))) _Float16 v16h;
typedef __attribute__((ext_vector_type(8)))  _Float16 v8h;
typedef __attribute__((ext_vector_type(8)))  float    v8f;
typedef __attribute__((ext_vector_type(4)))  int      v4i;

#define D_MODEL 1024
#define N_HEADS 16
#define D_K     64
#define BATCH   2
#define SEQ     2048
#define M_TOTAL (BATCH * SEQ)

// ---- CDNA5 async global->LDS copy ------------------------------------------
// Probe-learned signature: (global v4i*, shared v4i*, imm offset, imm cpol).
#if __has_builtin(__builtin_amdgcn_global_load_async_to_lds_b128)
#define USE_ASYNC_LDS 1
__device__ __forceinline__ void async_b128(void* lds, const void* g) {
    __builtin_amdgcn_global_load_async_to_lds_b128(
        (__attribute__((address_space(1))) v4i*)g,
        (__attribute__((address_space(3))) v4i*)lds,
        /*offset=*/0, /*cpol=*/0);
}
#else
#define USE_ASYNC_LDS 0
#endif

__device__ __forceinline__ void wait_async0() {
#if USE_ASYNC_LDS
#if __has_builtin(__builtin_amdgcn_s_wait_asynccnt)
    __builtin_amdgcn_s_wait_asynccnt(0);
#else
    asm volatile("s_wait_asynccnt 0x0" ::: "memory");
#endif
#endif
}

// 16 bytes global->LDS: async DMA when available, else synchronous.
__device__ __forceinline__ void copy_b128(void* lds, const void* g) {
#if USE_ASYNC_LDS
    async_b128(lds, g);
#else
    *(v8h*)lds = *(const v8h*)g;
#endif
}

// ---- WMMA fragment loaders (f16, 16x16x32) ---------------------------------
// A (16x32, MxK): lane = row (lane&15); lanes 0-15: K={0..7,16..23},
// lanes 16-31: K={8..15,24..31}.
__device__ __forceinline__ v16h load_frag_a(const _Float16* rowbase, int lane) {
    const int off = (lane >= 16) ? 8 : 0;
    v8h lo = *(const v8h*)(rowbase + off);
    v8h hi = *(const v8h*)(rowbase + 16 + off);
    v16h r;
#pragma unroll
    for (int i = 0; i < 8; ++i) { r[i] = lo[i]; r[i + 8] = hi[i]; }
    return r;
}
// B (32x16, KxN) stored transposed Bt[n][k]: lane = col (lane&15);
// lanes 0-15: K={0..7,16..23}, lanes 16-31: K={8..15,24..31}.
__device__ __forceinline__ v16h load_frag_b(const _Float16* rowbase, int lane) {
    const int off = (lane >= 16) ? 8 : 0;
    v8h lo = *(const v8h*)(rowbase + off);
    v8h hi = *(const v8h*)(rowbase + 16 + off);
    v16h r;
#pragma unroll
    for (int i = 0; i < 8; ++i) { r[i] = lo[i]; r[i + 8] = hi[i]; }
    return r;
}

// ---- One-time conversion kernels -------------------------------------------
__global__ __launch_bounds__(256) void cvt_f16(const float* __restrict__ in,
                                               _Float16* __restrict__ out) {
    const int i = (blockIdx.x * 256 + threadIdx.x) * 8;
#pragma unroll
    for (int j = 0; j < 8; ++j) out[i + j] = (_Float16)in[i + j];
}

// Wt[n][k] = (f16) W[k][n] via 32x32 LDS tile (pad 33 to dodge bank conflicts)
__global__ __launch_bounds__(256) void cvt_transpose(const float* __restrict__ W,
                                                     _Float16* __restrict__ Wt) {
    __shared__ _Float16 Ts[32][33];
    const int k0 = blockIdx.x * 32, n0 = blockIdx.y * 32;
    const int r  = threadIdx.x >> 3;
    const int c4 = (threadIdx.x & 7) * 4;
    const float* src = W + (size_t)(k0 + r) * D_MODEL + n0 + c4;
#pragma unroll
    for (int i = 0; i < 4; ++i) Ts[r][c4 + i] = (_Float16)src[i];
    __syncthreads();
    _Float16* dst = Wt + (size_t)(n0 + r) * D_MODEL + k0 + c4;
#pragma unroll
    for (int i = 0; i < 4; ++i) dst[i] = Ts[c4 + i][r];
}

// ---- GEMM: Y = (X @ W + bias) * scale, f16 inputs --------------------------
// X: [M,1024] f16 row-major; Wt: [1024,1024] f16, Wt[n][k] = W[k][n].
// Block: 256 thr (8 waves), tile 128x128, wave tile 32x64, K-step 32,
// double-buffered LDS with async global->LDS DMA.
template <int HEAD_OUT>
__global__ __launch_bounds__(256) void gemm_f16(const _Float16* __restrict__ X,
                                                const _Float16* __restrict__ Wt,
                                                const float* __restrict__ bias,
                                                void* __restrict__ Y,
                                                float scale) {
    __shared__ _Float16 As[2][128][32];
    __shared__ _Float16 Bs[2][128][32];

    const int tid  = threadIdx.x;
    const int lane = tid & 31;
    const int ln   = lane & 15;
    const int wave = tid >> 5;   // 0..7
    const int wm   = wave & 3;   // M sub-block (32 rows each)
    const int wn   = wave >> 2;  // N sub-block (64 cols each)
    const int gm0  = blockIdx.x * 128;
    const int gn0  = blockIdx.y * 128;

    const int crow = tid >> 1;          // copy row 0..127
    const int ccol = (tid & 1) * 16;    // two b128 transfers per matrix

    auto issue_tile = [&](int kt, int buf) {
        const _Float16* xa = X  + (size_t)(gm0 + crow) * D_MODEL + kt * 32 + ccol;
        const _Float16* wb = Wt + (size_t)(gn0 + crow) * D_MODEL + kt * 32 + ccol;
        copy_b128(&As[buf][crow][ccol],     xa);
        copy_b128(&As[buf][crow][ccol + 8], xa + 8);
        copy_b128(&Bs[buf][crow][ccol],     wb);
        copy_b128(&Bs[buf][crow][ccol + 8], wb + 8);
    };

    v8f acc[2][4] = {};
    issue_tile(0, 0);
    for (int kt = 0; kt < D_MODEL / 32; ++kt) {
        const int cur = kt & 1;
        wait_async0();        // this wave's DMA for tile kt has landed
        __syncthreads();      // everyone's tile kt visible; prev reads done
        if (kt + 1 < D_MODEL / 32) issue_tile(kt + 1, cur ^ 1);  // overlap DMA

        v16h a[2], b[4];
#pragma unroll
        for (int mi = 0; mi < 2; ++mi)
            a[mi] = load_frag_a(&As[cur][wm * 32 + mi * 16 + ln][0], lane);
#pragma unroll
        for (int ni = 0; ni < 4; ++ni)
            b[ni] = load_frag_b(&Bs[cur][wn * 64 + ni * 16 + ln][0], lane);
#pragma unroll
        for (int mi = 0; mi < 2; ++mi)
#pragma unroll
            for (int ni = 0; ni < 4; ++ni)
                acc[mi][ni] = __builtin_amdgcn_wmma_f32_16x16x32_f16(
                    false, a[mi], false, b[ni], (short)0, acc[mi][ni], false, false);
    }

    // Epilogue: C/D layout -> VGPR r holds rows r (lanes 0-15) / r+8 (16-31).
#pragma unroll
    for (int mi = 0; mi < 2; ++mi)
#pragma unroll
        for (int ni = 0; ni < 4; ++ni) {
            const int gn   = gn0 + wn * 64 + ni * 16 + ln;
            const float bv = bias[gn];
#pragma unroll
            for (int r = 0; r < 8; ++r) {
                const int gm    = gm0 + wm * 32 + mi * 16 + r + ((lane >= 16) ? 8 : 0);
                const float val = (acc[mi][ni][r] + bv) * scale;
                if (HEAD_OUT) {   // f16 [B, H, L, dk]
                    const int b_ = gm / SEQ, l_ = gm % SEQ;
                    const int h_ = gn / D_K, d_ = gn % D_K;
                    ((_Float16*)Y)[(((size_t)(b_ * N_HEADS + h_)) * SEQ + l_) * D_K + d_] =
                        (_Float16)val;
                } else {          // f32 [M, N] (final output)
                    ((float*)Y)[(size_t)gm * D_MODEL + gn] = val;
                }
            }
        }
}

// ---- Flash attention, f16 q/k/v in [B,H,L,dk], f16 ctx in [B,L,D] ----------
// Block 128 thr (4 waves): 64 query rows, KV streamed in 64-row tiles.
__global__ __launch_bounds__(128) void flash_attn(const _Float16* __restrict__ q,
                                                  const _Float16* __restrict__ k,
                                                  const _Float16* __restrict__ v,
                                                  _Float16* __restrict__ ctx) {
    __shared__ _Float16 Qs[64][64];
    __shared__ _Float16 Ks[64][64];       // natural [j][d] = Bt for S = Q K^T
    __shared__ _Float16 Vt[64][64];       // [d][j]          = Bt for O = P V
    __shared__ _Float16 Ps[4][16][64];    // per-wave P tile

    const int tid  = threadIdx.x;
    const int lane = tid & 31;
    const int wave = tid >> 5;
    const int ln   = lane & 15;
    const int q0   = blockIdx.x * 64;
    const int bh   = blockIdx.y;
    const int b_   = bh / N_HEADS;
    const int h_   = bh % N_HEADS;
    const size_t base = (size_t)bh * SEQ * D_K;

    {   // Q block: 512 b128 transfers, 4 per thread
        const int f0 = tid * 4;
#pragma unroll
        for (int t = 0; t < 4; ++t) {
            const int f = f0 + t, row = f >> 3, colh = (f & 7) * 8;
            copy_b128(&Qs[row][colh], q + base + (size_t)(q0 + row) * D_K + colh);
        }
    }

    float rm[8], rl[8];
    v8f   o[4] = {};
#pragma unroll
    for (int r = 0; r < 8; ++r) { rm[r] = -INFINITY; rl[r] = 0.0f; }

    for (int j0 = 0; j0 < SEQ; j0 += 64) {
        __syncthreads();   // previous tile fully consumed
        {   // K tile [64][64], natural layout
            const int f0 = tid * 4;
#pragma unroll
            for (int t = 0; t < 4; ++t) {
                const int f = f0 + t, row = f >> 3, colh = (f & 7) * 8;
                copy_b128(&Ks[row][colh], k + base + (size_t)(j0 + row) * D_K + colh);
            }
        }
        {   // V tile transposed -> Vt[d][j], packed b32 stores (2 j per store)
            const int jp = tid >> 2;              // row pair 0..31
            const int c  = (tid & 3) * 16;
            const _Float16* s0 = v + base + (size_t)(j0 + 2 * jp) * D_K + c;
            const _Float16* s1 = s0 + D_K;
            v8h a0 = *(const v8h*)s0, a1 = *(const v8h*)(s0 + 8);
            v8h b0 = *(const v8h*)s1, b1 = *(const v8h*)(s1 + 8);
#pragma unroll
            for (int i = 0; i < 8; ++i) {
                union { _Float16 h[2]; unsigned u; } p;
                p.h[0] = a0[i]; p.h[1] = b0[i];
                *(unsigned*)&Vt[c + i][2 * jp] = p.u;
                p.h[0] = a1[i]; p.h[1] = b1[i];
                *(unsigned*)&Vt[c + 8 + i][2 * jp] = p.u;
            }
        }
        wait_async0();
        __syncthreads();

        // S = Q_wave(16 rows) @ K_tile^T : 16x64, f32 accum
        v8f s[4] = {};
#pragma unroll
        for (int ks = 0; ks < 2; ++ks) {
            v16h a = load_frag_a(&Qs[wave * 16 + ln][ks * 32], lane);
#pragma unroll
            for (int ni = 0; ni < 4; ++ni) {
                v16h b = load_frag_b(&Ks[ni * 16 + ln][ks * 32], lane);
                s[ni] = __builtin_amdgcn_wmma_f32_16x16x32_f16(
                    false, a, false, b, (short)0, s[ni], false, false);
            }
        }

        // Online softmax: rows r / r+8 reduce independently (xor masks 1..8)
#pragma unroll
        for (int r = 0; r < 8; ++r) {
            float mx = fmaxf(fmaxf(s[0][r], s[1][r]), fmaxf(s[2][r], s[3][r]));
#pragma unroll
            for (int m = 8; m >= 1; m >>= 1) mx = fmaxf(mx, __shfl_xor(mx, m, 32));
            const float mnew  = fmaxf(rm[r], mx);
            const float alpha = __expf(rm[r] - mnew);
            float p[4], rs = 0.0f;
#pragma unroll
            for (int t = 0; t < 4; ++t) { p[t] = __expf(s[t][r] - mnew); rs += p[t]; }
#pragma unroll
            for (int m = 8; m >= 1; m >>= 1) rs += __shfl_xor(rs, m, 32);
            rl[r] = rl[r] * alpha + rs;
            rm[r] = mnew;
#pragma unroll
            for (int t = 0; t < 4; ++t) o[t][r] *= alpha;
            const int prow = r + ((lane >= 16) ? 8 : 0);
#pragma unroll
            for (int t = 0; t < 4; ++t)
                Ps[wave][prow][t * 16 + ln] = (_Float16)p[t];
        }

        // O += P @ V_tile (same-wave LDS ops are in-order; Ps is per-wave)
#pragma unroll
        for (int ks = 0; ks < 2; ++ks) {
            v16h a = load_frag_a(&Ps[wave][ln][ks * 32], lane);
#pragma unroll
            for (int t = 0; t < 4; ++t) {
                v16h b = load_frag_b(&Vt[t * 16 + ln][ks * 32], lane);
                o[t] = __builtin_amdgcn_wmma_f32_16x16x32_f16(
                    false, a, false, b, (short)0, o[t], false, false);
            }
        }
    }

    // ctx[b, l, h*64+d] = O / rowsum   (f16 for the output projection)
#pragma unroll
    for (int t = 0; t < 4; ++t) {
        const int d = h_ * D_K + t * 16 + ln;
#pragma unroll
        for (int r = 0; r < 8; ++r) {
            const int gq = q0 + wave * 16 + r + ((lane >= 16) ? 8 : 0);
            ctx[((size_t)(b_ * SEQ + gq)) * D_MODEL + d] = (_Float16)(o[t][r] / rl[r]);
        }
    }
}

// ---- Host-side launch ------------------------------------------------------
extern "C" void kernel_launch(void* const* d_in, const int* in_sizes, int n_in,
                              void* d_out, int out_size, void* d_ws, size_t ws_size,
                              hipStream_t stream) {
    (void)in_sizes; (void)n_in; (void)out_size; (void)ws_size;
    const float* Q  = (const float*)d_in[0];
    const float* K  = (const float*)d_in[1];
    const float* V  = (const float*)d_in[2];
    const float* Wq = (const float*)d_in[3];
    const float* bq = (const float*)d_in[4];
    const float* Wk = (const float*)d_in[5];
    const float* bk = (const float*)d_in[6];
    const float* Wv = (const float*)d_in[7];
    const float* bv = (const float*)d_in[8];
    const float* Wo = (const float*)d_in[9];
    const float* bo = (const float*)d_in[10];
    float* out = (float*)d_out;

    const size_t QKV = (size_t)BATCH * SEQ * D_MODEL;   // 4M elements
    _Float16* qh  = (_Float16*)d_ws;   // f16 q [B,H,L,dk]
    _Float16* kh  = qh  + QKV;
    _Float16* vh  = kh  + QKV;
    _Float16* xh  = vh  + QKV;         // converted input (reused Q,K,V)
    _Float16* cxh = xh  + QKV;         // ctx f16 [B,L,D]
    _Float16* wt  = cxh + QKV;         // transposed f16 weight (reused x4)
                                       // total 42 MB

    const dim3 gcv(QKV / 2048);                     // cvt_f16 grid
    const dim3 gwt(D_MODEL / 32, D_MODEL / 32);     // transpose grid
    const dim3 gg(M_TOTAL / 128, D_MODEL / 128);    // gemm grid
    const dim3 ga(SEQ / 64, BATCH * N_HEADS);       // attention grid
    const float qscale = 0.125f;                    // 1/sqrt(d_k)

    cvt_f16<<<gcv, 256, 0, stream>>>(Q, xh);
    cvt_transpose<<<gwt, 256, 0, stream>>>(Wq, wt);
    gemm_f16<1><<<gg, 256, 0, stream>>>(xh, wt, bq, qh, qscale);

    cvt_f16<<<gcv, 256, 0, stream>>>(K, xh);
    cvt_transpose<<<gwt, 256, 0, stream>>>(Wk, wt);
    gemm_f16<1><<<gg, 256, 0, stream>>>(xh, wt, bk, kh, 1.0f);

    cvt_f16<<<gcv, 256, 0, stream>>>(V, xh);
    cvt_transpose<<<gwt, 256, 0, stream>>>(Wv, wt);
    gemm_f16<1><<<gg, 256, 0, stream>>>(xh, wt, bv, vh, 1.0f);

    flash_attn<<<ga, 128, 0, stream>>>(qh, kh, vh, cxh);

    cvt_transpose<<<gwt, 256, 0, stream>>>(Wo, wt);
    gemm_f16<0><<<gg, 256, 0, stream>>>(cxh, wt, bo, out, 1.0f);
}
